// RGCN_link_predictor_61220463837501
// MI455X (gfx1250) — matrix operand order repaired and checked by reading, not compiled, not run on hardware
//
#include <hip/hip_runtime.h>

#define N_NODES 50000
#define NE 200000
#define F 128

typedef __attribute__((ext_vector_type(16))) __bf16        v16bf;
typedef __attribute__((ext_vector_type(8)))  float         v8f;
typedef __attribute__((ext_vector_type(8)))  unsigned int  v8u;

static __device__ __forceinline__ unsigned int f2bf(float f) {
  unsigned int u = __float_as_uint(f);
  unsigned int lsb = (u >> 16) & 1u;
  u += 0x7fffu + lsb;          // round-to-nearest-even
  return u >> 16;
}
static __device__ __forceinline__ unsigned int pack2(float lo, float hi) {
  return f2bf(lo) | (f2bf(hi) << 16);
}

// Pack fp32 weights [R][128][128] (k-major, col contiguous) into this file's
// per-lane B-fragment layout (contiguous k, same order as A) for v_wmma_f32_16x16x32_bf16:
// out[((r*4+kt)*8+ot)*32*8 + lane*8 + j] = {bf16 W[k][col], bf16 W[k+1][col]}
// with k = kt*32 + (lane>>4)*16 + 2j, col = ot*16 + (lane&15).
__global__ void pack_w_kernel(const float* __restrict__ W,
                              unsigned int* __restrict__ out, int total) {
  int idx = blockIdx.x * 256 + threadIdx.x;
  if (idx >= total) return;
  int j    = idx & 7;
  int lane = (idx >> 3) & 31;
  int ot   = (idx >> 8) & 7;
  int ktr  = idx >> 11;
  int kt   = ktr & 3;
  int r    = ktr >> 2;
  int col  = ot * 16 + (lane & 15);
  int k    = kt * 32 + (lane >> 4) * 16 + 2 * j;
  const float* Wr = W + (size_t)r * F * F;
  out[idx] = pack2(Wr[k * F + col], Wr[(k + 1) * F + col]);
}

__global__ void degree_kernel(const int* __restrict__ dst, float* __restrict__ deg) {
  int r = blockIdx.y;
  int e = blockIdx.x * 256 + threadIdx.x;
  if (e < NE) atomicAdd(&deg[r * N_NODES + dst[r * NE + e]], 1.0f);
}

// One wave per edge: coalesced 512B row gather, 4 f32 atomics per lane.
__global__ void scatter_kernel(const float* __restrict__ hin, const int* __restrict__ src,
                               const int* __restrict__ dst, float* __restrict__ agg) {
  int r = blockIdx.y;
  int wave = threadIdx.x >> 5, lane = threadIdx.x & 31;
  int e = blockIdx.x * 8 + wave;
  if (e >= NE) return;
  int s = src[r * NE + e], d = dst[r * NE + e];
  float4 v = ((const float4*)(hin + (size_t)s * F))[lane];
  float* o = agg + ((size_t)r * N_NODES + d) * F + lane * 4;
  atomicAdd(o + 0, v.x);
  atomicAdd(o + 1, v.y);
  atomicAdd(o + 2, v.z);
  atomicAdd(o + 3, v.w);
}

// Per wave: 16 nodes x 128 out, K=128, 4 relations summed (ReLU per relation
// for layer 1). A staged as normalized bf16 in LDS; B read from packed weights.
__global__ void __launch_bounds__(128)
layer_kernel(const float* __restrict__ agg, const float* __restrict__ deg,
             const unsigned int* __restrict__ Wp, const float* __restrict__ bias,
             float* __restrict__ hout, int do_relu) {
  __shared__ unsigned int lds[4][16 * 64];   // 4 waves x (16 rows x 128 bf16)
  int wave = threadIdx.x >> 5, lane = threadIdx.x & 31;
  int tile = blockIdx.x * 4 + wave;
  if (tile > (N_NODES / 16 - 1)) tile = N_NODES / 16 - 1;   // dup-write tail, benign
  int nodeBase = tile * 16;
  int m = lane & 15, khalf = lane >> 4;
  unsigned int* t_lds = lds[wave];

  v8f hsum[8];
#pragma unroll
  for (int t = 0; t < 8; ++t)
#pragma unroll
    for (int i = 0; i < 8; ++i) hsum[t][i] = 0.0f;

  for (int r = 0; r < 4; ++r) {
    // stage normalized bf16 A tile (16 x 128)
#pragma unroll 4
    for (int mm = 0; mm < 16; ++mm) {
      int node = nodeBase + mm;
      float dg = deg[r * N_NODES + node];
      float nrm = dg > 0.0f ? 1.0f / fmaxf(dg, 1.0f) : 0.0f;
      float4 v = ((const float4*)(agg + ((size_t)r * N_NODES + node) * F))[lane];
      unsigned int p0 = pack2(v.x * nrm, v.y * nrm);
      unsigned int p1 = pack2(v.z * nrm, v.w * nrm);
      ((uint2*)t_lds)[mm * 32 + lane] = make_uint2(p0, p1);
    }
    __syncthreads();

    v8f acc[8];
#pragma unroll
    for (int t = 0; t < 8; ++t)
#pragma unroll
      for (int i = 0; i < 8; ++i) acc[t][i] = 0.0f;

#pragma unroll
    for (int kt = 0; kt < 4; ++kt) {
      const uint4* ap = (const uint4*)(t_lds + m * 64 + kt * 16 + khalf * 8);
      uint4 a0 = ap[0], a1 = ap[1];
      v8u au = {a0.x, a0.y, a0.z, a0.w, a1.x, a1.y, a1.z, a1.w};
      v16bf afrag = __builtin_bit_cast(v16bf, au);
#pragma unroll
      for (int t = 0; t < 8; ++t) {
        const uint4* bp = (const uint4*)(Wp + ((size_t)(((r * 4 + kt) * 8 + t) * 32 + lane) * 8));
        uint4 b0 = bp[0], b1 = bp[1];
        v8u bu = {b0.x, b0.y, b0.z, b0.w, b1.x, b1.y, b1.z, b1.w};
        v16bf bfrag = __builtin_bit_cast(v16bf, bu);
        acc[t] = __builtin_amdgcn_wmma_f32_16x16x32_bf16(
            false, afrag, false, bfrag, (short)0, acc[t], false, false);
      }
    }
    __syncthreads();

#pragma unroll
    for (int t = 0; t < 8; ++t) {
      float bv = bias[r * F + t * 16 + m];
#pragma unroll
      for (int i = 0; i < 8; ++i) {
        float o = acc[t][i] + bv;
        if (do_relu) o = fmaxf(o, 0.0f);
        hsum[t][i] += o;
      }
    }
  }

  // C layout: VGPR i -> row khalf*8+i, col = t*16 + (lane&15)
#pragma unroll
  for (int t = 0; t < 8; ++t)
#pragma unroll
    for (int i = 0; i < 8; ++i)
      hout[(size_t)(nodeBase + khalf * 8 + i) * F + t * 16 + m] = hsum[t][i];
}

// Per wave: 16 edges. e = h[src]*h[dst] staged bf16 in LDS; WMMA vs packed
// P1w; ReLU; 128->2 linear folded into sigmoid(l1-l0) via shfl reduction.
__global__ void __launch_bounds__(128)
edge_kernel(const float* __restrict__ h,
            const unsigned int* __restrict__ P1p, const float* __restrict__ P1b,
            const float* __restrict__ P2w, const float* __restrict__ P2b,
            const int* __restrict__ pos_src, const int* __restrict__ pos_dst,
            const int* __restrict__ neg_src, const int* __restrict__ neg_dst,
            float* __restrict__ out) {
  __shared__ unsigned int lds[4][16 * 64];
  int wave = threadIdx.x >> 5, lane = threadIdx.x & 31;
  int rel = blockIdx.y;                 // 0..3 pos, 4..7 neg
  int rr = rel & 3;
  const int* es = (rel >= 4 ? neg_src : pos_src) + (size_t)rr * NE;
  const int* ed = (rel >= 4 ? neg_dst : pos_dst) + (size_t)rr * NE;
  float* outp = out + (size_t)rel * NE; // [pos(4xNE), neg(4xNE)] flat
  int ebase = (blockIdx.x * 4 + wave) * 16;
  int m = lane & 15, khalf = lane >> 4;
  unsigned int* t_lds = lds[wave];

#pragma unroll 4
  for (int mm = 0; mm < 16; ++mm) {
    int e = ebase + mm;
    int s = es[e], d = ed[e];
    float4 a = ((const float4*)(h + (size_t)s * F))[lane];
    float4 b = ((const float4*)(h + (size_t)d * F))[lane];
    unsigned int p0 = pack2(a.x * b.x, a.y * b.y);
    unsigned int p1 = pack2(a.z * b.z, a.w * b.w);
    ((uint2*)t_lds)[mm * 32 + lane] = make_uint2(p0, p1);
  }
  __syncthreads();

  v8f acc[8];
#pragma unroll
  for (int t = 0; t < 8; ++t)
#pragma unroll
    for (int i = 0; i < 8; ++i) acc[t][i] = 0.0f;

#pragma unroll
  for (int kt = 0; kt < 4; ++kt) {
    const uint4* ap = (const uint4*)(t_lds + m * 64 + kt * 16 + khalf * 8);
    uint4 a0 = ap[0], a1 = ap[1];
    v8u au = {a0.x, a0.y, a0.z, a0.w, a1.x, a1.y, a1.z, a1.w};
    v16bf afrag = __builtin_bit_cast(v16bf, au);
#pragma unroll
    for (int t = 0; t < 8; ++t) {
      const uint4* bp = (const uint4*)(P1p + ((size_t)((kt * 8 + t) * 32 + lane) * 8));
      uint4 b0 = bp[0], b1 = bp[1];
      v8u bu = {b0.x, b0.y, b0.z, b0.w, b1.x, b1.y, b1.z, b1.w};
      v16bf bfrag = __builtin_bit_cast(v16bf, bu);
      acc[t] = __builtin_amdgcn_wmma_f32_16x16x32_bf16(
          false, afrag, false, bfrag, (short)0, acc[t], false, false);
    }
  }

  // bias + ReLU, then partial dot with (P2w[:,1]-P2w[:,0])
  float part[8];
#pragma unroll
  for (int i = 0; i < 8; ++i) part[i] = 0.0f;
#pragma unroll
  for (int t = 0; t < 8; ++t) {
    float bv = P1b[t * 16 + m];
    float2 w = ((const float2*)P2w)[t * 16 + m];
    float dw = w.y - w.x;
#pragma unroll
    for (int i = 0; i < 8; ++i) {
      float z = fmaxf(acc[t][i] + bv, 0.0f);
      part[i] += z * dw;
    }
  }
  // reduce across the 16 lanes of each half-wave
#pragma unroll
  for (int off = 1; off < 16; off <<= 1)
#pragma unroll
    for (int i = 0; i < 8; ++i) part[i] += __shfl_xor(part[i], off, 32);

  float db = P2b[1] - P2b[0];
  if (m == 0) {
#pragma unroll
    for (int i = 0; i < 8; ++i) {
      float d = part[i] + db;
      outp[ebase + khalf * 8 + i] = 1.0f / (1.0f + __expf(-d));
    }
  }
}

extern "C" void kernel_launch(void* const* d_in, const int* in_sizes, int n_in,
                              void* d_out, int out_size, void* d_ws, size_t ws_size,
                              hipStream_t stream) {
  const float* x   = (const float*)d_in[0];
  const float* W1  = (const float*)d_in[1];
  const float* b1  = (const float*)d_in[2];
  const float* W2  = (const float*)d_in[3];
  const float* b2  = (const float*)d_in[4];
  const float* P1w = (const float*)d_in[5];
  const float* P1b = (const float*)d_in[6];
  const float* P2w = (const float*)d_in[7];
  const float* P2b = (const float*)d_in[8];
  const int* src     = (const int*)d_in[9];
  const int* dst     = (const int*)d_in[10];
  const int* pos_src = (const int*)d_in[11];
  const int* pos_dst = (const int*)d_in[12];
  const int* neg_src = (const int*)d_in[13];
  const int* neg_dst = (const int*)d_in[14];
  (void)in_sizes; (void)n_in; (void)out_size; (void)ws_size;

  char* ws = (char*)d_ws;
  size_t off = 0;
  auto walloc = [&](size_t bytes) -> void* {
    void* p = ws + off;
    off += (bytes + 255) & ~(size_t)255;
    return p;
  };
  const size_t AGG_BYTES = (size_t)4 * N_NODES * F * sizeof(float);
  float* deg = (float*)walloc((size_t)4 * N_NODES * sizeof(float));
  float* agg = (float*)walloc(AGG_BYTES);
  float* h1  = (float*)walloc((size_t)N_NODES * F * sizeof(float));
  float* h2  = (float*)walloc((size_t)N_NODES * F * sizeof(float));
  unsigned int* W1p = (unsigned int*)walloc(32768 * sizeof(unsigned int));
  unsigned int* W2p = (unsigned int*)walloc(32768 * sizeof(unsigned int));
  unsigned int* P1p = (unsigned int*)walloc(8192 * sizeof(unsigned int));

  hipMemsetAsync(deg, 0, (size_t)4 * N_NODES * sizeof(float), stream);
  hipMemsetAsync(agg, 0, AGG_BYTES, stream);

  pack_w_kernel<<<(32768 + 255) / 256, 256, 0, stream>>>(W1, W1p, 32768);
  pack_w_kernel<<<(32768 + 255) / 256, 256, 0, stream>>>(W2, W2p, 32768);
  pack_w_kernel<<<(8192 + 255) / 256, 256, 0, stream>>>(P1w, P1p, 8192);

  degree_kernel<<<dim3((NE + 255) / 256, 4), 256, 0, stream>>>(dst, deg);

  scatter_kernel<<<dim3(NE / 8, 4), 256, 0, stream>>>(x, src, dst, agg);
  layer_kernel<<<(N_NODES / 16 + 3) / 4, 128, 0, stream>>>(agg, deg, W1p, b1, h1, 1);

  hipMemsetAsync(agg, 0, AGG_BYTES, stream);
  scatter_kernel<<<dim3(NE / 8, 4), 256, 0, stream>>>(h1, src, dst, agg);
  layer_kernel<<<(N_NODES / 16 + 3) / 4, 128, 0, stream>>>(agg, deg, W2p, b2, h2, 0);

  edge_kernel<<<dim3(NE / 16 / 4, 8), 128, 0, stream>>>(
      h2, P1p, P1b, P2w, P2b, pos_src, pos_dst, neg_src, neg_dst, (float*)d_out);
}
